// BucketAdjustedHinge_29626684408053
// MI455X (gfx1250) — matrix operand order repaired and checked by reading, not compiled, not run on hardware
//
#include <hip/hip_runtime.h>
#include <hip/hip_bf16.h>

// CDNA5 / gfx1250: wave32, WMMA 16x16x32 f16 -> f32
typedef __attribute__((ext_vector_type(16))) _Float16 v16h;
typedef __attribute__((ext_vector_type(8)))  float    v8f;

#define BLOCK_THREADS    256
#define WAVES_PER_BLOCK  8           // 256 / wave32
#define ROW_STRIDE       18          // 16 cols + 2 pad: 8*18 = 144 ≡ 16 (mod 64) -> no LDS bank conflicts

__device__ __forceinline__ float softplus_f(float v) {
    // stable softplus, matches jax.nn.softplus
    return fmaxf(v, 0.f) + log1pf(expf(-fabsf(v)));
}

// Evaluate one 16-sample tile: D[i,j] = Σk min(x01_i, baseknot_k)*wbase_k
//                                      + Σk min(x01_i, adjknot_k)*wadj_{j,k}
//                                      + base_bias + adj_bias_j            (via constant-1 row K=16)
__device__ __forceinline__ v8f tile_eval(float x01,
                                         v16h aKbase, v16h aKadj, v16h aFix,
                                         v16h bWbase, v16h bWadj) {
    const _Float16 xh = (_Float16)x01;
    v16h xv;
    #pragma unroll
    for (int e = 0; e < 16; ++e) xv[e] = xh;                 // splat
    v16h abase = __builtin_elementwise_min(xv, aKbase);      // v_pk_min_num_f16 x8
    v16h madj  = __builtin_elementwise_min(xv, aKadj);
    // rows K>=16 of A_adj are constants (1 at K=16 for bias row, else 0)
    v16h aadj  = __builtin_shufflevector(madj, aFix,
                                         0, 1, 2, 3, 4, 5, 6, 7,
                                         24, 25, 26, 27, 28, 29, 30, 31);
    v8f d = {};
    d = __builtin_amdgcn_wmma_f32_16x16x32_f16(false, abase, false, bWbase, (short)0, d, false, false);
    d = __builtin_amdgcn_wmma_f32_16x16x32_f16(false, aadj,  false, bWadj,  (short)0, d, false, false);
    return d;
}

__global__ __launch_bounds__(BLOCK_THREADS)
void BucketAdjustedHinge_wmma_kernel(const float* __restrict__ X,
                                     const float* __restrict__ x_mins,
                                     const float* __restrict__ x_maxs,
                                     const float* __restrict__ clip_los,
                                     const float* __restrict__ clip_his,
                                     const float* __restrict__ base_knots,   // [32]
                                     const float* __restrict__ base_raw_w,   // [32]
                                     const float* __restrict__ base_bias,    // [1]
                                     const float* __restrict__ adj_knots,    // [16]
                                     const float* __restrict__ adj_raw_w,    // [16,16]
                                     const float* __restrict__ adj_bias,     // [16]
                                     const unsigned int* __restrict__ bidx_lo, // int64 viewed as u32 pairs
                                     float* __restrict__ out,
                                     int n)
{
    __shared__ float4 params[16];                                  // (min, inv_scale, lo', hi')
    __shared__ float  dstage[WAVES_PER_BLOCK][2][16 * ROW_STRIDE]; // per-wave D staging, padded rows

    if (threadIdx.x < 16) {
        const int t = threadIdx.x;
        const float mn = x_mins[t], mx = x_maxs[t];
        float lo = clip_los[t], hi = clip_his[t];
        lo = __builtin_isfinite(lo) ? lo : -__builtin_inff();      // non-finite bound => no clipping
        hi = __builtin_isfinite(hi) ? hi :  __builtin_inff();
        params[t] = make_float4(mn, 1.f / (mx - mn + 1e-12f), lo, hi);
    }
    __syncthreads();

    const int lane = threadIdx.x & 31;
    const int hf   = lane >> 4;       // 0: lanes 0-15, 1: lanes 16-31
    const int col  = lane & 15;
    const int wv   = threadIdx.x >> 5;

    // ---- loop-invariant per-lane WMMA operands (ISA VGPR layouts) ----
    // A (16-bit, 16x32): element e of lane -> K = e + 8*(e>=8) + 8*hf
    // B (16-bit, 32x16): lane = column (col), element e -> K = e + 8*(e>=8) + 8*hf
    const float bb = base_bias[0];
    v16h aKbase, aKadj, aFix, bWbase, bWadj;
    #pragma unroll
    for (int e = 0; e < 16; ++e) {
        const int Ka = e + ((e >= 8) ? 8 : 0) + 8 * hf;
        const int Kb = e + ((e >= 8) ? 8 : 0) + 8 * hf;
        aKbase[e] = (_Float16)base_knots[Ka];
        bWbase[e] = (_Float16)softplus_f(base_raw_w[Kb]);          // every column identical
        float wadj;
        if (Kb < 16)       wadj = softplus_f(adj_raw_w[col * 16 + Kb]); // bucket 'col' weights
        else if (Kb == 16) wadj = bb + adj_bias[col];              // bias row
        else               wadj = 0.f;
        bWadj[e] = (_Float16)wadj;
        aKadj[e] = (e < 8) ? (_Float16)adj_knots[e + 8 * hf] : (_Float16)0.f;
        aFix[e]  = (_Float16)0.f;
    }
    aFix[8] = hf ? (_Float16)0.f : (_Float16)1.f;                  // A[K=16] = 1 (bias row)

    const int ntiles = (n + 31) >> 5;                              // 32 samples per wave-iteration
    const int gwave  = blockIdx.x * WAVES_PER_BLOCK + wv;
    const int nwaves = gridDim.x * WAVES_PER_BLOCK;
    float* stage0 = &dstage[wv][0][0];
    float* stage1 = &dstage[wv][1][0];

    for (int t = gwave; t < ntiles; t += nwaves) {
        const long long s  = (long long)t * 32 + lane;             // own sample
        const long long sl = (s < n) ? s : (long long)(n - 1);     // clamped: keep EXEC full

        const float        xraw = X[sl];
        const unsigned int b    = bidx_lo[2 * sl] & 15u;           // low dword of int64 index

        // prefetch next grid-stride tile (global_prefetch_b8)
        const long long sp = s + (long long)nwaves * 32;
        if (sp < n) {
            __builtin_prefetch(&X[sp], 0, 0);
            __builtin_prefetch(&bidx_lo[2 * sp], 0, 0);
        }

        const float4 p  = params[b];
        const float  xc = fminf(fmaxf(xraw, p.z), p.w);
        const float  x01 = fminf(fmaxf((xc - p.x) * p.y, 0.f), 1.f);

        // mirror x01 across lane halves: tile0 = samples of lanes 0-15, tile1 = lanes 16-31
        const float x01o  = __shfl_xor(x01, 16);
        const float x01t0 = hf ? x01o : x01;
        const float x01t1 = hf ? x01  : x01o;

        const v8f d0 = tile_eval(x01t0, aKbase, aKadj, aFix, bWbase, bWadj);
        const v8f d1 = tile_eval(x01t1, aKbase, aKadj, aFix, bWbase, bWadj);

        // stage D tiles to LDS (C/D layout: lane holds rows r+8*hf of column 'col')
        #pragma unroll
        for (int r = 0; r < 8; ++r) {
            stage0[(r + 8 * hf) * ROW_STRIDE + col] = d0[r];
            stage1[(r + 8 * hf) * ROW_STRIDE + col] = d1[r];
        }
        // each lane extracts its own sample's bucket column: D[row=col][j=b] of tile 'hf'
        float* mystage = hf ? stage1 : stage0;                     // intra-wave LDS ops are in-order
        const float res = mystage[col * ROW_STRIDE + (int)b];

        if (s < n) out[s] = res;
    }
}

extern "C" void kernel_launch(void* const* d_in, const int* in_sizes, int n_in,
                              void* d_out, int out_size, void* d_ws, size_t ws_size,
                              hipStream_t stream) {
    const float* X          = (const float*)d_in[0];
    const float* x_mins     = (const float*)d_in[1];
    const float* x_maxs     = (const float*)d_in[2];
    const float* clip_los   = (const float*)d_in[3];
    const float* clip_his   = (const float*)d_in[4];
    const float* base_knots = (const float*)d_in[5];
    const float* base_raw_w = (const float*)d_in[6];
    const float* base_bias  = (const float*)d_in[7];
    const float* adj_knots  = (const float*)d_in[8];
    const float* adj_raw_w  = (const float*)d_in[9];
    const float* adj_bias   = (const float*)d_in[10];
    const unsigned int* bidx = (const unsigned int*)d_in[11];   // int64 -> u32 pairs (LE)
    float* out = (float*)d_out;

    const int n = in_sizes[0];
    const int ntiles = (n + 31) / 32;
    int blocks = (ntiles + WAVES_PER_BLOCK - 1) / WAVES_PER_BLOCK;
    if (blocks > 2048) blocks = 2048;   // grid-stride; ~16K waves saturates HBM
    if (blocks < 1) blocks = 1;

    hipLaunchKernelGGL(BucketAdjustedHinge_wmma_kernel,
                       dim3(blocks), dim3(BLOCK_THREADS), 0, stream,
                       X, x_mins, x_maxs, clip_los, clip_his,
                       base_knots, base_raw_w, base_bias,
                       adj_knots, adj_raw_w, adj_bias,
                       bidx, out, n);
}
